// DeltaRuleMemory_86878598463928
// MI455X (gfx1250) — matrix operand order repaired and checked by reading, not compiled, not run on hardware
//
#include <hip/hip_runtime.h>
#include <hip/hip_bf16.h>
#include <math.h>

#define CB  2
#define CT  2048
#define CDM 1024
#define CNH 16
#define CHD 64
#define CINNER 1024
#define CM  (CB * CT)   // 4096 rows

typedef __attribute__((ext_vector_type(16))) _Float16 v16h;
typedef __attribute__((ext_vector_type(8)))  _Float16 v8h;
typedef __attribute__((ext_vector_type(8)))  float    v8f;

// ---- WMMA fragment helpers (contiguous k order k = 16*half + s, same for A and B) ----------------
// A-matrix 16x32 f16: lane l (<16) = row l, holds K = half*16+{0..15}
__device__ __forceinline__ v16h load_a_frag(const _Float16* p) {
  v8h lo = *(const v8h*)(p);
  v8h hi = *(const v8h*)(p + 8);
  v16h r;
#pragma unroll
  for (int i = 0; i < 8; ++i) { r[i] = lo[i]; r[8 + i] = hi[i]; }
  return r;
}
// B-matrix 32x16 f16: lane n = column n; lanes<16 hold K=0..15, lanes>=16 hold K=16..31
__device__ __forceinline__ v16h load_b_frag(const _Float16* p) {
  return *(const v16h*)(p);
}
__device__ __forceinline__ v8f wmma_f16(v16h a, v16h b, v8f c) {
  return __builtin_amdgcn_wmma_f32_16x16x32_f16(false, a, false, b, (short)0, c,
                                                false, false);
}

// ---- elementwise f32 -> f16 ---------------------------------------------
__global__ void cvt_f32_f16(const float* __restrict__ src,
                            _Float16* __restrict__ dst, int n) {
  int i = blockIdx.x * blockDim.x + threadIdx.x;
  if (i < n) dst[i] = (_Float16)src[i];
}

// ---- C[m][n] = sum_k A[m][k] * W[n][k]  (A:[M,K] f16, W:[N,K] f16) -------
__global__ void __launch_bounds__(128)
gemm_xwT_wmma(const _Float16* __restrict__ A, const _Float16* __restrict__ W,
              float* __restrict__ C, int M, int N, int K) {
  const int lane = threadIdx.x & 31;
  const int wid  = threadIdx.x >> 5;
  const int half = lane >> 4;
  const int l    = lane & 15;
  const int m0 = (blockIdx.y * 4 + wid) * 16;
  const int n0 = blockIdx.x * 64;
  if (m0 >= M || n0 >= N) return;
  const _Float16* arow = A + (size_t)(m0 + l) * K + half * 16;
  const _Float16* w0 = W + (size_t)(n0 +  0 + l) * K + half * 16;
  const _Float16* w1 = W + (size_t)(n0 + 16 + l) * K + half * 16;
  const _Float16* w2 = W + (size_t)(n0 + 32 + l) * K + half * 16;
  const _Float16* w3 = W + (size_t)(n0 + 48 + l) * K + half * 16;
  v8f a0 = {}, a1 = {}, a2 = {}, a3 = {};
  for (int k0 = 0; k0 < K; k0 += 32) {
    __builtin_prefetch(arow + k0 + 256, 0, 1);   // global_prefetch_b8
    v16h a = load_a_frag(arow + k0);
    a0 = wmma_f16(a, load_b_frag(w0 + k0), a0);
    a1 = wmma_f16(a, load_b_frag(w1 + k0), a1);
    a2 = wmma_f16(a, load_b_frag(w2 + k0), a2);
    a3 = wmma_f16(a, load_b_frag(w3 + k0), a3);
  }
  // C/D layout: VGPR r, lanes<16 -> row r, lanes>=16 -> row r+8; col = n-tile + l
  float* crow = C + (size_t)(m0 + 8 * half) * N + n0 + l;
#pragma unroll
  for (int r = 0; r < 8; ++r) {
    crow[(size_t)r * N +  0] = a0[r];
    crow[(size_t)r * N + 16] = a1[r];
    crow[(size_t)r * N + 32] = a2[r];
    crow[(size_t)r * N + 48] = a3[r];
  }
}

// ---- RoPE q->f16, RoPE k in place (f32), reduce sum|k'|, sum|v| ----------
__global__ void rope_reduce(const float* __restrict__ qf, float* __restrict__ kf,
                            const float* __restrict__ vf,
                            _Float16* __restrict__ qh, float* __restrict__ sums) {
  const int n = CB * CT * CNH * (CHD / 2);
  int idx = blockIdx.x * blockDim.x + threadIdx.x;
  float ak = 0.f, av = 0.f;
  if (idx < n) {
    int dd  = idx & 31;           // rotary pair index (0..31)
    int h   = (idx >> 5) & (CNH - 1);
    int row = idx >> 9;           // b*T + t
    int t   = row & (CT - 1);
    // inv_freq = 10000^(-2*dd/64)
    float inv = __expf(-((float)(2 * dd) / (float)CHD) * 9.210340371976184f);
    float fr = (float)t * inv;
    float c = cosf(fr), s = sinf(fr);
    size_t base = (size_t)row * CINNER + h * CHD + dd;
    float q1 = qf[base], q2 = qf[base + 32];
    qh[base]      = (_Float16)(q1 * c - q2 * s);
    qh[base + 32] = (_Float16)(q2 * c + q1 * s);
    float k1 = kf[base], k2 = kf[base + 32];
    float k1r = k1 * c - k2 * s;
    float k2r = k2 * c + k1 * s;
    kf[base] = k1r; kf[base + 32] = k2r;
    ak = fabsf(k1r) + fabsf(k2r);
    av = fabsf(vf[base]) + fabsf(vf[base + 32]);
  }
#pragma unroll
  for (int off = 16; off > 0; off >>= 1) {
    ak += __shfl_down(ak, off, 32);
    av += __shfl_down(av, off, 32);
  }
  if ((threadIdx.x & 31) == 0) {
    atomicAdd(&sums[0], ak);
    atomicAdd(&sums[1], av);
  }
}

// ---- ternary quantization of k (row-major, global threshold) -------------
__global__ void quant_k(const float* __restrict__ kf, _Float16* __restrict__ kq,
                        const float* __restrict__ sums, int n) {
  int i = blockIdx.x * blockDim.x + threadIdx.x;
  if (i >= n) return;
  float thrk = fminf(fmaxf(sums[0] / (float)n, 0.01f), 10.f);
  float k = kf[i];
  kq[i] = (_Float16)((k > thrk) ? 1.f : (k < -thrk) ? -1.f : 0.f);
}

// ---- ternary quantize v + transpose to vT[(b,h,d)][t] --------------------
// Makes the attention B-fragment for V a single contiguous 32B load.
__global__ void __launch_bounds__(256)
quant_v_T(const float* __restrict__ vf, _Float16* __restrict__ vqT,
          const float* __restrict__ sums) {
  __shared__ _Float16 tile[32][33];
  const int b  = blockIdx.z;
  const int t0 = blockIdx.x * 32;
  const int c0 = blockIdx.y * 32;          // c = h*64 + d
  const int n  = CM * CINNER;
  float thrv = fminf(fmaxf(sums[1] / (float)n, 0.01f), 10.f);
  const int tx = threadIdx.x, ty = threadIdx.y;
#pragma unroll
  for (int r = 0; r < 4; ++r) {
    int t = t0 + ty + r * 8;
    float v = vf[(size_t)(b * CT + t) * CINNER + c0 + tx];
    tile[ty + r * 8][tx] = (_Float16)((v > thrv) ? 1.f : (v < -thrv) ? -1.f : 0.f);
  }
  __syncthreads();
#pragma unroll
  for (int r = 0; r < 4; ++r) {
    int c = c0 + ty + r * 8;
    int t = t0 + tx;
    vqT[(size_t)(b * CINNER + c) * CT + t] = tile[tx][ty + r * 8];
  }
}

// ---- beta[b,t,h] = sigmoid(x . Wb[h] + bb[h]) ---------------------------
__global__ void beta_gate(const float* __restrict__ x, const float* __restrict__ Wb,
                          const float* __restrict__ bb, float* __restrict__ beta) {
  int h   = blockIdx.x & (CNH - 1);
  int row = blockIdx.x >> 4;
  const float* xr = x  + (size_t)row * CDM;
  const float* wr = Wb + (size_t)h   * CDM;
  float p = 0.f;
  for (int k = threadIdx.x; k < CDM; k += 64) p += xr[k] * wr[k];
  __shared__ float red[64];
  red[threadIdx.x] = p;
  __syncthreads();
#pragma unroll
  for (int s = 32; s > 0; s >>= 1) {
    if (threadIdx.x < s) red[threadIdx.x] += red[threadIdx.x + s];
    __syncthreads();
  }
  if (threadIdx.x == 0)
    beta[(size_t)row * CNH + h] = 1.f / (1.f + __expf(-(red[0] + bb[h])));
}

// ---- attention: one wave per (b, h, 16-row i-tile) ----------------------
// out[b,i,h,:] = sum_{j<=i} (q_i.k_j) * alpha^(i-j) * beta_j * v_j
// Output written directly as f16 (feeds the Wo GEMM).
__global__ void __launch_bounds__(32)
attn_wmma(const _Float16* __restrict__ qh, const _Float16* __restrict__ kq,
          const _Float16* __restrict__ vqT, const float* __restrict__ beta,
          const float* __restrict__ alpha_log, _Float16* __restrict__ outh) {
  const int lane = threadIdx.x;
  const int half = lane >> 4;
  const int l    = lane & 15;
  const int NIT  = CT / 16;
  int it = blockIdx.x % NIT;
  int h  = (blockIdx.x / NIT) % CNH;
  int b  = blockIdx.x / (NIT * CNH);
  const int i0 = it * 16;
  const int imax = i0 + 15;
  float alpha = 1.f / (1.f + __expf(-alpha_log[h]));
  float la = __logf(fmaxf(alpha, 1e-6f));

  // q A-fragments for d=[0,32) and d=[32,64)
  size_t qrow = (size_t)(b * CT + i0 + l) * CINNER + h * CHD;
  v16h qa0 = load_a_frag(qh + qrow + 0  + half * 16);
  v16h qa1 = load_a_frag(qh + qrow + 32 + half * 16);

  // transposed V base for this (b,h): vT[d][t], row stride CT
  const _Float16* vbase = vqT + (size_t)(b * CINNER + h * CHD) * CT;

  v8f o[4] = {};                   // 16x16 f32 tiles covering d = 0..63
  __shared__ _Float16 sS[16 * 32]; // staged S' tile (16 i x 32 j), row-major
  const int irow = i0 + 8 * half;

  for (int j0 = 0; j0 <= imax; j0 += 32) {
#pragma unroll
    for (int sub = 0; sub < 2; ++sub) {
      int js = j0 + sub * 16;
      v8f s = {};
      if (js <= imax) {
        size_t krow = (size_t)(b * CT + js + l) * CINNER + h * CHD;
        __builtin_prefetch(kq + krow + (size_t)32 * CINNER, 0, 1);
        v16h kb0 = load_b_frag(kq + krow + 0  + half * 16); // B[d][j]=k[j][d]
        v16h kb1 = load_b_frag(kq + krow + 32 + half * 16);
        s = wmma_f16(qa0, kb0, s);
        s = wmma_f16(qa1, kb1, s);
        int j = js + l;                         // this lane's column
        float bj = beta[(size_t)(b * CT + j) * CNH + h];
#pragma unroll
        for (int r = 0; r < 8; ++r) {
          int i = irow + r;
          float f = (j <= i) ? __expf((float)(i - j) * la) * bj : 0.f;
          s[r] *= f;
        }
      }
#pragma unroll
      for (int r = 0; r < 8; ++r)
        sS[(r + 8 * half) * 32 + sub * 16 + l] = (_Float16)s[r];
    }
    __syncthreads(); // single wave: DS in-order; acts as compiler fence

    // Re-gather S' in A-layout: lane row = l, K (=j_rel) = half*16 + e, contiguous (same order as V)
    v16h sa;
#pragma unroll
    for (int e = 0; e < 8; ++e) {
      sa[e]     = sS[l * 32 + half * 16 + e];
      sa[8 + e] = sS[l * 32 + half * 16 + 8 + e];
    }
    // V B-fragments from transposed layout: lane = column d, K = contiguous j
    const _Float16* vj = vbase + (size_t)l * CT + j0 + half * 16;
#pragma unroll
    for (int nt = 0; nt < 4; ++nt) {
      v16h vb = load_b_frag(vj + (size_t)(nt * 16) * CT);
      o[nt] = wmma_f16(sa, vb, o[nt]);
    }
    __syncthreads();
  }

  _Float16* ob = outh + (size_t)(b * CT + i0 + 8 * half) * CINNER + h * CHD + l;
#pragma unroll
  for (int r = 0; r < 8; ++r) {
    ob[(size_t)r * CINNER +  0] = (_Float16)o[0][r];
    ob[(size_t)r * CINNER + 16] = (_Float16)o[1][r];
    ob[(size_t)r * CINNER + 32] = (_Float16)o[2][r];
    ob[(size_t)r * CINNER + 48] = (_Float16)o[3][r];
  }
}

extern "C" void kernel_launch(void* const* d_in, const int* in_sizes, int n_in,
                              void* d_out, int out_size, void* d_ws, size_t ws_size,
                              hipStream_t stream) {
  const float* x  = (const float*)d_in[0];
  const float* Wq = (const float*)d_in[1];
  const float* Wk = (const float*)d_in[2];
  const float* Wv = (const float*)d_in[3];
  const float* Wo = (const float*)d_in[4];
  const float* Wb = (const float*)d_in[5];
  const float* bb = (const float*)d_in[6];
  const float* al = (const float*)d_in[7];
  float* out = (float*)d_out;

  char* w = (char*)d_ws;
  const size_t MB = 1ull << 20;
  _Float16* xh   = (_Float16*)(w);            // 8 MB   x in f16
  _Float16* WqH  = (_Float16*)(w +  8 * MB);  // 2 MB
  _Float16* WkH  = (_Float16*)(w + 10 * MB);  // 2 MB
  _Float16* WvH  = (_Float16*)(w + 12 * MB);  // 2 MB
  _Float16* WoH  = (_Float16*)(w + 14 * MB);  // 2 MB
  float*    qf   = (float*)   (w + 16 * MB);  // 16 MB  q fp32 (dead after rope)
  float*    kf   = (float*)   (w + 32 * MB);  // 16 MB  k fp32 (dead after quant)
  float*    vf   = (float*)   (w + 48 * MB);  // 16 MB  v fp32
  _Float16* qh16 = (_Float16*)(w + 64 * MB);  // 8 MB   roped q f16
  _Float16* kq16 = (_Float16*)(w + 72 * MB);  // 8 MB   ternary k f16 (row-major)
  _Float16* vqT  = (_Float16*)(w + 80 * MB);  // 8 MB   ternary v f16 (transposed)
  float*    betaB= (float*)   (w + 88 * MB);  // 256 KB
  float*    sums = (float*)   (w + 88 * MB + 512 * 1024); // 8 B
  _Float16* attnh = (_Float16*)qf;  // reuse q fp32 region: attention output f16

  hipMemsetAsync(sums, 0, 2 * sizeof(float), stream);

  const int nx = CM * CDM;           // 4096*1024
  const int nw = CINNER * CDM;       // 1024*1024
  cvt_f32_f16<<<(nx + 255) / 256, 256, 0, stream>>>(x,  xh,  nx);
  cvt_f32_f16<<<(nw + 255) / 256, 256, 0, stream>>>(Wq, WqH, nw);
  cvt_f32_f16<<<(nw + 255) / 256, 256, 0, stream>>>(Wk, WkH, nw);
  cvt_f32_f16<<<(nw + 255) / 256, 256, 0, stream>>>(Wv, WvH, nw);
  cvt_f32_f16<<<(nw + 255) / 256, 256, 0, stream>>>(Wo, WoH, nw);

  dim3 ggrid(CINNER / 64, CM / 64);  // (16, 64), 4 waves/block, 16 rows each
  gemm_xwT_wmma<<<ggrid, 128, 0, stream>>>(xh, WqH, qf, CM, CINNER, CDM);
  gemm_xwT_wmma<<<ggrid, 128, 0, stream>>>(xh, WkH, kf, CM, CINNER, CDM);
  gemm_xwT_wmma<<<ggrid, 128, 0, stream>>>(xh, WvH, vf, CM, CINNER, CDM);

  const int nr = CB * CT * CNH * (CHD / 2);  // 2,097,152 rotary pairs
  rope_reduce<<<(nr + 255) / 256, 256, 0, stream>>>(qf, kf, vf, qh16, sums);

  const int nq = CM * CINNER;        // 4,194,304
  quant_k<<<(nq + 255) / 256, 256, 0, stream>>>(kf, kq16, sums, nq);
  quant_v_T<<<dim3(CT / 32, CINNER / 32, CB), dim3(32, 8), 0, stream>>>(vf, vqT, sums);

  beta_gate<<<CM * CNH, 64, 0, stream>>>(x, Wb, bb, betaB);

  attn_wmma<<<CB * CNH * (CT / 16), 32, 0, stream>>>(qh16, kq16, vqT, betaB, al, attnh);

  gemm_xwT_wmma<<<ggrid, 128, 0, stream>>>(attnh, WoH, out, CM, CDM, CINNER);
}
